// attention_27633819583077
// MI455X (gfx1250) — matrix operand order repaired and checked by reading, not compiled, not run on hardware
//
#include <hip/hip_runtime.h>
#include <hip/hip_bf16.h>
#include <math.h>

typedef __attribute__((ext_vector_type(16))) __bf16    v16bf;
typedef __attribute__((ext_vector_type(8)))  float     v8f;
typedef __attribute__((ext_vector_type(4)))  float     v4f;
typedef __attribute__((ext_vector_type(8)))  unsigned  v8u;

// global (addrspace 1) pointer type: survives asm laundering with its
// address space intact -> loads stay global_load_b128, not flat_load.
typedef __attribute__((address_space(1))) const v16bf* gv16p;

#define NITER 8
#define ROWS_PER_BLOCK (128 * NITER)

__device__ __forceinline__ __bf16 f2bf(float f) { return static_cast<__bf16>(f); }

// single-op f32 relu (avoid fmaxf's canonicalize+max pair)
__device__ __forceinline__ float relu1(float x) {
  float r;
  asm("v_max_num_f32 %0, 0, %1" : "=v"(r) : "v"(x));
  return r;
}

// packed bf16 relu: two elements per op
__device__ __forceinline__ unsigned pkrelu(unsigned x) {
  unsigned r;
  asm("v_pk_max_num_bf16 %0, %1, 0" : "=v"(r) : "v"(x));
  return r;
}

__device__ __forceinline__ void sm_combine(float& m, float& l, float m2, float l2) {
  float M = fmaxf(m, m2);
  if (M == -__builtin_inff()) { m = M; l = 0.f; return; }
  l = l * __expf(m - M) + l2 * __expf(m2 - M);
  m = M;
}

// ---------------------------------------------------------------------------
// K0: fold feature2 into bias c1; swizzle W1a into WMMA A-operand layout and
//     W2^T into WMMA B-operand layout (bf16), W1a in contiguous k order, W2^T in ISA table k order.
// ---------------------------------------------------------------------------
__global__ void prep_kernel(const float* __restrict__ W1, const float* __restrict__ b1,
                            const float* __restrict__ W2, const float* __restrict__ f2,
                            __bf16* __restrict__ w1op, __bf16* __restrict__ w2op,
                            float* __restrict__ c1) {
  int t = threadIdx.x;
  if (t < 64) {
    float acc = b1[t];
    for (int k = 0; k < 64; ++k) acc += W1[t * 128 + 64 + k] * f2[k];
    c1[t] = acc;
  }
  // A-operand (16x32 bf16): lane<16: M=lane, K=0..15 ; lane>=16: M=lane-16, K=16..31
  for (int idx = t; idx < 4096; idx += 256) {
    int e = idx & 15, lane = (idx >> 4) & 31, jt = (idx >> 9) & 3, kb = idx >> 11;
    int half = lane >> 4;
    int j = jt * 16 + (lane & 15);
    int koff = half * 16 + e;
    w1op[idx] = f2bf(W1[j * 128 + kb * 32 + koff]);
  }
  // B-operand (32x16 bf16): lane<16: N=lane, K = e<8 ? e : 16+(e-8) ; lane>=16: N=lane-16, K = e<8 ? 8+e : 24+(e-8)
  for (int idx = t; idx < 4096; idx += 256) {
    int e = idx & 15, lane = (idx >> 4) & 31, nt = (idx >> 9) & 3, kb = idx >> 11;
    int j = nt * 16 + (lane & 15);              // B[k][j] = W2[j][k]
    int k = kb * 32 + ((e < 8) ? ((lane >> 4) * 8 + e) : (16 + (lane >> 4) * 8 + (e - 8)));
    w2op[idx] = f2bf(W2[j * 64 + k]);
  }
}

// ---------------------------------------------------------------------------
// K1: fused MLP: h1^T = relu(W1a x f1^T + c1) ; h2 = relu(h1 x W2^T + b2) ;
//     s = h2 . w3 + b3 -> scores + per-block online-softmax stats.
// Weight pointers (addrspace(1)) are laundered per iteration so LICM cannot
// re-hoist the operand loads into ~128 resident VGPRs; they stay L0-hot.
// ---------------------------------------------------------------------------
__global__ void __launch_bounds__(256)
mlp_kernel(const float* __restrict__ f1, const float* __restrict__ b2,
           const float* __restrict__ W3, const float* __restrict__ b3,
           const __bf16* __restrict__ w1op, const __bf16* __restrict__ w2op,
           const float* __restrict__ c1, float* __restrict__ scores,
           float* __restrict__ bmax, float* __restrict__ bsum, int N) {
  const int lane = threadIdx.x & 31;
  const int wave = threadIdx.x >> 5;
  const int half = lane >> 4;
  const int l16  = lane & 15;

  gv16p w1g = (gv16p)(const v16bf*)w1op;
  gv16p w2g = (gv16p)(const v16bf*)w2op;

  v8f c1v[4];
#pragma unroll
  for (int jt = 0; jt < 4; ++jt) c1v[jt] = *(const v8f*)(c1 + jt * 16 + half * 8);
  float b2v[4], w3v[4];
#pragma unroll
  for (int nt = 0; nt < 4; ++nt) {
    b2v[nt] = b2[nt * 16 + l16];
    w3v[nt] = W3[nt * 16 + l16];
  }
  const float bias3 = b3[0];

  // after the width-halving reduce, lane owns row element:
  const int e_own  = 4 * (lane & 1) + 2 * ((lane >> 1) & 1) + ((lane >> 2) & 1);
  const bool owner = (lane & 8) == 0;

  float lm = -__builtin_inff(), ll = 0.f;

  for (int it = 0; it < NITER; ++it) {
    // opaque, loop-carried copies: defeats LICM/CSE of the weight loads
    asm volatile("" : "+s"(w1g), "+s"(w2g));

    const int row0 = blockIdx.x * ROWS_PER_BLOCK + it * 128 + wave * 16;
    // ---- load 16 feature rows as B operands (64B contiguous per lane) ----
    int r = row0 + l16; if (r > N - 1) r = N - 1;
    const float* rp = f1 + (size_t)r * 64 + half * 16;
    v16bf fb[2];
#pragma unroll
    for (int kb = 0; kb < 2; ++kb) {
      const v4f* p = (const v4f*)(rp + kb * 32);
#pragma unroll
      for (int q = 0; q < 4; ++q) {
        v4f v = p[q];
#pragma unroll
        for (int x = 0; x < 4; ++x) fb[kb][q * 4 + x] = f2bf(v[x]);
      }
    }
    // ---- layer 1 (transposed): h1^T tiles, M=feature, N=row ----
    v8f h[4];
#pragma unroll
    for (int jt = 0; jt < 4; ++jt) {
      v8f a = c1v[jt];
      a = __builtin_amdgcn_wmma_f32_16x16x32_bf16(false, w1g[jt * 32 + lane], false,
                                                  fb[0], (short)0, a, false, false);
      a = __builtin_amdgcn_wmma_f32_16x16x32_bf16(false, w1g[(4 + jt) * 32 + lane], false,
                                                  fb[1], (short)0, a, false, false);
      h[jt] = a;
    }
    // ---- convert then packed-bf16 relu (relu commutes with RNE rounding) ----
    v16bf a2[2];
#pragma unroll
    for (int e = 0; e < 8; ++e) {
      a2[0][e]     = f2bf(h[0][e]);
      a2[0][8 + e] = f2bf(h[1][e]);
      a2[1][e]     = f2bf(h[2][e]);
      a2[1][8 + e] = f2bf(h[3][e]);
    }
#pragma unroll
    for (int o = 0; o < 2; ++o) {
      v8u u = __builtin_bit_cast(v8u, a2[o]);
#pragma unroll
      for (int i = 0; i < 8; ++i) u[i] = pkrelu(u[i]);
      a2[o] = __builtin_bit_cast(v16bf, u);
    }
    // ---- layer 2 + layer-3 weighted sum ----
    v8f part;
#pragma unroll
    for (int e = 0; e < 8; ++e) part[e] = 0.f;
#pragma unroll
    for (int nt = 0; nt < 4; ++nt) {
      v8f g;
#pragma unroll
      for (int e = 0; e < 8; ++e) g[e] = b2v[nt];
      g = __builtin_amdgcn_wmma_f32_16x16x32_bf16(false, a2[0], false,
                                                  w2g[nt * 32 + lane], (short)0, g, false, false);
      g = __builtin_amdgcn_wmma_f32_16x16x32_bf16(false, a2[1], false,
                                                  w2g[(4 + nt) * 32 + lane], (short)0, g, false, false);
#pragma unroll
      for (int e = 0; e < 8; ++e) part[e] = fmaf(relu1(g[e]), w3v[nt], part[e]);
    }
    // ---- width-halving cross-lane reduce: 8 shuffles instead of 32 ----
    float q4[4];
#pragma unroll
    for (int i = 0; i < 4; ++i) {
      float send = (lane & 1) ? part[i] : part[i + 4];
      float recv = __shfl_xor(send, 1, 32);
      q4[i] = ((lane & 1) ? part[i + 4] : part[i]) + recv;
    }
    float q2[2];
#pragma unroll
    for (int i = 0; i < 2; ++i) {
      float send = (lane & 2) ? q4[i] : q4[i + 2];
      float recv = __shfl_xor(send, 2, 32);
      q2[i] = ((lane & 2) ? q4[i + 2] : q4[i]) + recv;
    }
    {
      float send = (lane & 4) ? q2[0] : q2[1];
      float recv = __shfl_xor(send, 4, 32);
      float q1 = ((lane & 4) ? q2[1] : q2[0]) + recv;
      q1 += __shfl_xor(q1, 8, 32);
      if (owner) {
        int row = row0 + half * 8 + e_own;
        if (row < N) {
          float s = q1 + bias3;
          scores[row] = s;
          if (s > lm) { ll *= __expf(lm - s); lm = s; }
          ll += __expf(s - lm);
        }
      }
    }
  }

  // ---- block reduce (max, sumexp) ----
#pragma unroll
  for (int m = 16; m >= 1; m >>= 1) {
    float om = __shfl_xor(lm, m, 32);
    float ol = __shfl_xor(ll, m, 32);
    sm_combine(lm, ll, om, ol);
  }
  __shared__ float smx[8], slx[8];
  if (lane == 0) { smx[wave] = lm; slx[wave] = ll; }
  __syncthreads();
  if (threadIdx.x == 0) {
    float M = smx[0], L = slx[0];
    for (int w = 1; w < 8; ++w) sm_combine(M, L, smx[w], slx[w]);
    bmax[blockIdx.x] = M;
    bsum[blockIdx.x] = L;
  }
}

// ---------------------------------------------------------------------------
// K2: merge per-block stats -> global (M, L)
// ---------------------------------------------------------------------------
__global__ void reduce_kernel(const float* __restrict__ bmax, const float* __restrict__ bsum,
                              float* __restrict__ g, int nb) {
  float lm = -__builtin_inff(), ll = 0.f;
  for (int i = threadIdx.x; i < nb; i += 256) sm_combine(lm, ll, bmax[i], bsum[i]);
#pragma unroll
  for (int m = 16; m >= 1; m >>= 1) {
    float om = __shfl_xor(lm, m, 32);
    float ol = __shfl_xor(ll, m, 32);
    sm_combine(lm, ll, om, ol);
  }
  __shared__ float smx[8], slx[8];
  int lane = threadIdx.x & 31, wave = threadIdx.x >> 5;
  if (lane == 0) { smx[wave] = lm; slx[wave] = ll; }
  __syncthreads();
  if (threadIdx.x == 0) {
    float M = smx[0], L = slx[0];
    for (int w = 1; w < 8; ++w) sm_combine(M, L, smx[w], slx[w]);
    g[0] = M;
    g[1] = L;
  }
}

// ---------------------------------------------------------------------------
// K3: in-place normalize: out[i] = exp(s - M) / L
// ---------------------------------------------------------------------------
__global__ void norm_kernel(float* __restrict__ out, const float* __restrict__ g, int N) {
  int i = blockIdx.x * 256 + threadIdx.x;
  if (i < N) out[i] = __expf(out[i] - g[0]) / g[1];
}

extern "C" void kernel_launch(void* const* d_in, const int* in_sizes, int n_in,
                              void* d_out, int out_size, void* d_ws, size_t ws_size,
                              hipStream_t stream) {
  const float* f1 = (const float*)d_in[0];
  const float* f2 = (const float*)d_in[1];
  const float* W1 = (const float*)d_in[2];
  const float* b1 = (const float*)d_in[3];
  const float* W2 = (const float*)d_in[4];
  const float* b2 = (const float*)d_in[5];
  const float* W3 = (const float*)d_in[6];
  const float* b3 = (const float*)d_in[7];
  const int N = in_sizes[0] / 64;
  float* out = (float*)d_out;

  char* ws = (char*)d_ws;
  __bf16* w1op = (__bf16*)(ws);            // 8 KB
  __bf16* w2op = (__bf16*)(ws + 8192);     // 8 KB
  float*  c1   = (float*)(ws + 16384);     // 256 B
  float*  g    = (float*)(ws + 16640);     // M, L
  const int nblocks = (N + ROWS_PER_BLOCK - 1) / ROWS_PER_BLOCK;
  float* bmax = (float*)(ws + 16704);
  float* bsum = bmax + nblocks;

  prep_kernel<<<1, 256, 0, stream>>>(W1, b1, W2, f2, w1op, w2op, c1);
  mlp_kernel<<<nblocks, 256, 0, stream>>>(f1, b2, W3, b3, w1op, w2op, c1,
                                          out, bmax, bsum, N);
  reduce_kernel<<<1, 256, 0, stream>>>(bmax, bsum, g, nblocks);
  norm_kernel<<<(N + 255) / 256, 256, 0, stream>>>(out, g, N);
}
